// YatAttention_55886114455726
// MI455X (gfx1250) — matrix operand order repaired and checked by reading, not compiled, not run on hardware
//
#include <hip/hip_runtime.h>
#include <math.h>

// ---------------- problem constants ----------------
#define E_DIM   1024
#define H_HEADS 16
#define D_HEAD  64
#define B_BATCH 2
#define S_SEQ   2048
#define BS_ROWS (B_BATCH * S_SEQ)      // 4096
#define BH_CNT  (B_BATCH * H_HEADS)    // 32
#define YAT_EPS 0.1f

typedef __attribute__((ext_vector_type(16))) __bf16 v16bf;
typedef __attribute__((ext_vector_type(8)))  float  v8f;
typedef __attribute__((ext_vector_type(4)))  int    v4i;

// ---------------- optional gfx1250 async global->LDS path ----------------
#if defined(__has_builtin)
#if __has_builtin(__builtin_amdgcn_global_load_async_to_lds_b128)
#define HAVE_ASYNC_LDS 1
#endif
#endif

// copy 16 bytes global -> LDS (async when available)
__device__ __forceinline__ void g2l_b128(unsigned short* lds,
                                         const unsigned short* g) {
#ifdef HAVE_ASYNC_LDS
    __builtin_amdgcn_global_load_async_to_lds_b128(
        (v4i*)g, (v4i*)lds, 0, 0);
#else
    *(uint4*)lds = *(const uint4*)g;
#endif
}
__device__ __forceinline__ void wait_async_lds() {
#ifdef HAVE_ASYNC_LDS
#if __has_builtin(__builtin_amdgcn_s_wait_asynccnt)
    __builtin_amdgcn_s_wait_asynccnt(0);
#else
    asm volatile("s_wait_asynccnt 0x0" ::: "memory");
#endif
#endif
}

// ---------------- bf16 helpers (bit-level, RNE) ----------------
__device__ __forceinline__ unsigned short f2bf(float f) {
    union { float f; unsigned int u; } c; c.f = f;
    unsigned int u = c.u;
    u += 0x7FFFu + ((u >> 16) & 1u);   // round-to-nearest-even
    return (unsigned short)(u >> 16);
}

// fragment <-> two 16B chunks.  ISA A-layout (16-bit 16x32): lane holds
// M=lane%16; elements 0..7 = K[8*half .. 8*half+7], 8..15 = K[16+8*half ..].
// B (32x16): N=lane%16. This code uses element e = K[16*half + e] (16 contiguous) for A and B.
union frag_u {
    uint4 v[2];
    v16bf f;
};

// ---------------- kernel 1: fp32 -> bf16 convert ----------------
__global__ void cvt_f32_to_bf16(const float* __restrict__ in,
                                unsigned short* __restrict__ out, int n) {
    int i = blockIdx.x * blockDim.x + threadIdx.x;
    int stride = gridDim.x * blockDim.x;
    for (; i < n; i += stride) out[i] = f2bf(in[i]);
}

// ---------------- kernel 2: bf16 WMMA GEMM (64x64 tile / 4-wave block) ------
// C[M,N](f32) = A[M,K](bf16 rm) @ B[K,N](bf16 rm) + bias[N]
// B is staged TRANSPOSED in LDS so every fragment is two aligned ds_load_b128.
// A fragments load straight from global (two b128 per wave per K-step).
__global__ __launch_bounds__(128)
void gemm_bf16_wmma(const unsigned short* __restrict__ A,
                    const unsigned short* __restrict__ Bw,
                    const float* __restrict__ bias,
                    float* __restrict__ C, int M, int N, int K) {
    __shared__ __align__(16) unsigned short btT[64 * 32];   // [n][k]
    const int tid  = threadIdx.x;
    const int lane = tid & 31;
    const int wv   = tid >> 5;                 // wave 0..3 -> M sub-tile
    const int row  = lane & 15, half = lane >> 4;
    const int tn = blockIdx.x * 64, tm = blockIdx.y * 64;
    const unsigned short* arow = A + (size_t)(tm + wv * 16 + row) * K;

    v8f acc[4] = {{}, {}, {}, {}};
    for (int k0 = 0; k0 < K; k0 += 32) {
        // stage B tile 32x64 transposed -> btT[n][k]
        #pragma unroll
        for (int j = 0; j < 2; ++j) {
            int f = tid + 128 * j;             // 0..255 uint4 chunks
            int r = f >> 3, c8 = f & 7;        // r = k row, c8 = 8-col chunk
            union { uint4 v; unsigned short s[8]; } u;
            u.v = *(const uint4*)(Bw + (size_t)(k0 + r) * N + tn + c8 * 8);
            #pragma unroll
            for (int jj = 0; jj < 8; ++jj)
                btT[(c8 * 8 + jj) * 32 + r] = u.s[jj];
        }
        __syncthreads();

        // A fragment: two contiguous 16B global loads
        frag_u ua;
        ua.v[0] = *(const uint4*)(arow + k0 + 16 * half);
        ua.v[1] = *(const uint4*)(arow + k0 + 16 * half + 8);
        if (k0 + 32 < K) __builtin_prefetch(arow + k0 + 32);

        // 4 N sub-tiles reuse the A fragment
        #pragma unroll
        for (int t = 0; t < 4; ++t) {
            frag_u ub;
            const uint4* p = (const uint4*)(btT + (t * 16 + row) * 32 + 16 * half);
            ub.v[0] = p[0];
            ub.v[1] = p[1];
            acc[t] = __builtin_amdgcn_wmma_f32_16x16x32_bf16(
                         false, ua.f, false, ub.f, (short)0, acc[t], false, false);
        }
        __syncthreads();
    }
    // C layout: VGPR r -> M = r + 8*half, N = lane%16
    #pragma unroll
    for (int t = 0; t < 4; ++t) {
        #pragma unroll
        for (int r = 0; r < 8; ++r) {
            int m = tm + wv * 16 + r + 8 * half;
            C[(size_t)m * N + tn + t * 16 + row] = acc[t][r] + bias[tn + t * 16 + row];
        }
    }
}

// ---------------- kernel 3: L2-normalize + reorder to [B*H, S, 64] bf16 -----
__global__ __launch_bounds__(32)
void norm_reorder(const float* __restrict__ src,      // [BS, E] fp32
                  unsigned short* __restrict__ dst,   // [BH, S, 64] bf16
                  int do_norm) {
    const int s  = blockIdx.x;
    const int bh = blockIdx.y;
    const int b  = bh / H_HEADS, h = bh % H_HEADS;
    const int lane = threadIdx.x;
    size_t srow = (size_t)(b * S_SEQ + s) * E_DIM + h * D_HEAD;
    float q0 = src[srow + lane];
    float q1 = src[srow + lane + 32];
    float inv = 1.0f;
    if (do_norm) {
        float ss = q0 * q0 + q1 * q1;
        #pragma unroll
        for (int m = 1; m < 32; m <<= 1) ss += __shfl_xor(ss, m, 32);
        inv = 1.0f / (sqrtf(ss) + 1e-8f);
    }
    size_t drow = ((size_t)bh * S_SEQ + s) * D_HEAD;
    dst[drow + lane]      = f2bf(q0 * inv);
    dst[drow + lane + 32] = f2bf(q1 * inv);
}

// ---------------- kernel 4: Yat attention (single pass, no softmax) ---------
// 4-wave block: 64 queries of one (b,h); waves share each 32-key K/V tile.
// O = (sum_k score*V) / (sum_k score + 1e-6), accumulated in one pass.
__global__ __launch_bounds__(128)
void yat_attn_wmma(const unsigned short* __restrict__ Qn,   // [BH,S,64] L2-normed
                   const unsigned short* __restrict__ Kn,   // [BH,S,64] L2-normed
                   const unsigned short* __restrict__ Vb,   // [BH,S,64]
                   unsigned short* __restrict__ Obf) {      // [BS, E] merged heads
    __shared__ __align__(16) unsigned short qtile[4][16 * 64];
    __shared__ __align__(16) unsigned short ktile[32 * 64];  // [key][d]
    __shared__ __align__(16) unsigned short vT[64 * 32];     // [d][key] (transposed)
    __shared__ __align__(16) unsigned short stile[4][16 * 32];

    const int tid  = threadIdx.x;
    const int lane = tid & 31;
    const int wv   = tid >> 5;
    const int row  = lane & 15, half = lane >> 4;
    const int bh = blockIdx.y;
    const int b  = bh / H_HEADS, h = bh % H_HEADS;
    const int qt0 = blockIdx.x * 64 + wv * 16;

    // each wave stages its own 16x64 Q tile (async when available)
    const unsigned short* qbase = Qn + ((size_t)bh * S_SEQ + qt0) * D_HEAD;
    #pragma unroll
    for (int j = 0; j < 4; ++j)
        g2l_b128(qtile[wv] + (lane + 32 * j) * 8, qbase + (size_t)(lane + 32 * j) * 8);
    wait_async_lds();
    __syncthreads();

    // Q A-fragments (K=64 -> two K=32 WMMAs), contiguous 16B chunks
    frag_u aq0, aq1;
    {
        const unsigned short* qr = qtile[wv] + row * 64;
        aq0.v[0] = *(const uint4*)(qr + 16 * half);
        aq0.v[1] = *(const uint4*)(qr + 16 * half + 8);
        aq1.v[0] = *(const uint4*)(qr + 32 + 16 * half);
        aq1.v[1] = *(const uint4*)(qr + 32 + 16 * half + 8);
    }

    v8f o[4] = {{}, {}, {}, {}};
    float rs[8] = {0.f, 0.f, 0.f, 0.f, 0.f, 0.f, 0.f, 0.f};

    for (int kt0 = 0; kt0 < S_SEQ; kt0 += 32) {
        const unsigned short* kbase = Kn + ((size_t)bh * S_SEQ + kt0) * D_HEAD;
        const unsigned short* vbase = Vb + ((size_t)bh * S_SEQ + kt0) * D_HEAD;
        // K tile: straight copy (async path), 256 chunks over 128 threads
        #pragma unroll
        for (int j = 0; j < 2; ++j) {
            int f = tid + 128 * j;
            g2l_b128(ktile + f * 8, kbase + (size_t)f * 8);
        }
        // V tile: transpose-on-store so S@V B-fragments become contiguous
        #pragma unroll
        for (int j = 0; j < 2; ++j) {
            int f = tid + 128 * j;
            int r = f >> 3, c8 = f & 7;
            union { uint4 v; unsigned short s[8]; } u;
            u.v = *(const uint4*)(vbase + (size_t)r * 64 + c8 * 8);
            #pragma unroll
            for (int jj = 0; jj < 8; ++jj)
                vT[(c8 * 8 + jj) * 32 + r] = u.s[jj];
        }
        wait_async_lds();
        __syncthreads();

        // two 16-key sub-tiles: Q.K^T -> score transform -> stash bf16 scores
        #pragma unroll
        for (int sub = 0; sub < 2; ++sub) {
            frag_u bk0, bk1;   // B = Kn^T; ktile is already N(key)-major
            const unsigned short* kr = ktile + (sub * 16 + row) * 64;
            bk0.v[0] = *(const uint4*)(kr + 16 * half);
            bk0.v[1] = *(const uint4*)(kr + 16 * half + 8);
            bk1.v[0] = *(const uint4*)(kr + 32 + 16 * half);
            bk1.v[1] = *(const uint4*)(kr + 32 + 16 * half + 8);
            v8f c = {};
            c = __builtin_amdgcn_wmma_f32_16x16x32_bf16(false, aq0.f, false, bk0.f,
                                                        (short)0, c, false, false);
            c = __builtin_amdgcn_wmma_f32_16x16x32_bf16(false, aq1.f, false, bk1.f,
                                                        (short)0, c, false, false);
            #pragma unroll
            for (int r = 0; r < 8; ++r) {
                float d  = c[r];
                float sc = (1.0f + d) * (1.0f + d) / (YAT_EPS + (1.0f - d));
                rs[r] += sc;
                // C-layout scatter: query row r+8*half, key col sub*16+lane%16
                stile[wv][(r + 8 * half) * 32 + sub * 16 + row] = f2bf(sc);
            }
        }
        // scores back as A-fragment (contiguous; wave-local LDS, DS is in-order)
        frag_u as;
        {
            const unsigned short* sr = stile[wv] + row * 32;
            as.v[0] = *(const uint4*)(sr + 16 * half);
            as.v[1] = *(const uint4*)(sr + 16 * half + 8);
        }
        // S @ V: four 16-dim slices from transposed V
        #pragma unroll
        for (int dt = 0; dt < 4; ++dt) {
            frag_u bv;
            const uint4* p = (const uint4*)(vT + (dt * 16 + row) * 32 + 16 * half);
            bv.v[0] = p[0];
            bv.v[1] = p[1];
            o[dt] = __builtin_amdgcn_wmma_f32_16x16x32_bf16(
                        false, as.f, false, bv.f, (short)0, o[dt], false, false);
        }
        __syncthreads();   // all waves done reading ktile/vT before restage
    }

    // rowsum: reduce across the 16 lanes sharing a query row
    #pragma unroll
    for (int r = 0; r < 8; ++r) {
        #pragma unroll
        for (int m = 1; m < 16; m <<= 1) rs[r] += __shfl_xor(rs[r], m, 32);
    }
    // normalize and write merged-head bf16 output [BS, E]
    #pragma unroll
    for (int r = 0; r < 8; ++r) {
        float inv = 1.0f / (rs[r] + 1e-6f);
        size_t orow = (size_t)(b * S_SEQ + qt0 + r + 8 * half) * E_DIM + h * D_HEAD;
        #pragma unroll
        for (int dt = 0; dt < 4; ++dt)
            Obf[orow + dt * 16 + row] = f2bf(o[dt][r] * inv);
    }
}

// ---------------- host-side launch ----------------
extern "C" void kernel_launch(void* const* d_in, const int* in_sizes, int n_in,
                              void* d_out, int out_size, void* d_ws, size_t ws_size,
                              hipStream_t stream) {
    const float* x  = (const float*)d_in[0];
    const float* Wq = (const float*)d_in[1];
    const float* bq = (const float*)d_in[2];
    const float* Wk = (const float*)d_in[3];
    const float* bk = (const float*)d_in[4];
    const float* Wv = (const float*)d_in[5];
    const float* bv = (const float*)d_in[6];
    const float* Wo = (const float*)d_in[7];
    const float* bo = (const float*)d_in[8];
    float* out = (float*)d_out;

    const size_t MB = 1024 * 1024;
    char* ws = (char*)d_ws;
    unsigned short* Xbf = (unsigned short*)(ws + 0 * MB);    // 8 MB  [BS,E]
    unsigned short* Wqb = (unsigned short*)(ws + 8 * MB);    // 2 MB each
    unsigned short* Wkb = (unsigned short*)(ws + 10 * MB);
    unsigned short* Wvb = (unsigned short*)(ws + 12 * MB);
    unsigned short* Wob = (unsigned short*)(ws + 14 * MB);
    float* Qf = (float*)(ws + 16 * MB);                      // 16 MB [BS,E]
    float* Kf = (float*)(ws + 32 * MB);
    float* Vf = (float*)(ws + 48 * MB);
    unsigned short* Qn  = (unsigned short*)(ws + 64 * MB);   // 8 MB [BH,S,64]
    unsigned short* Kn  = (unsigned short*)(ws + 72 * MB);
    unsigned short* Vbh = (unsigned short*)(ws + 80 * MB);
    unsigned short* Obf = (unsigned short*)(ws + 16 * MB);   // reuse Qf slot

    // 1) convert inputs to bf16
    cvt_f32_to_bf16<<<1024, 256, 0, stream>>>(x,  Xbf, BS_ROWS * E_DIM);
    cvt_f32_to_bf16<<<1024, 256, 0, stream>>>(Wq, Wqb, E_DIM * E_DIM);
    cvt_f32_to_bf16<<<1024, 256, 0, stream>>>(Wk, Wkb, E_DIM * E_DIM);
    cvt_f32_to_bf16<<<1024, 256, 0, stream>>>(Wv, Wvb, E_DIM * E_DIM);
    cvt_f32_to_bf16<<<1024, 256, 0, stream>>>(Wo, Wob, E_DIM * E_DIM);

    // 2) Q/K/V projections (bf16 WMMA, fp32 accum): 64x64 tile per 4-wave block
    dim3 gemm_grid(E_DIM / 64, BS_ROWS / 64);
    gemm_bf16_wmma<<<gemm_grid, 128, 0, stream>>>(Xbf, Wqb, bq, Qf, BS_ROWS, E_DIM, E_DIM);
    gemm_bf16_wmma<<<gemm_grid, 128, 0, stream>>>(Xbf, Wkb, bk, Kf, BS_ROWS, E_DIM, E_DIM);
    gemm_bf16_wmma<<<gemm_grid, 128, 0, stream>>>(Xbf, Wvb, bv, Vf, BS_ROWS, E_DIM, E_DIM);

    // 3) normalize Q/K, reorder V -> [BH,S,64] bf16
    dim3 ngrid(S_SEQ, BH_CNT);
    norm_reorder<<<ngrid, 32, 0, stream>>>(Qf, Qn, 1);
    norm_reorder<<<ngrid, 32, 0, stream>>>(Kf, Kn, 1);
    norm_reorder<<<ngrid, 32, 0, stream>>>(Vf, Vbh, 0);

    // 4) Yat attention (64 queries per block, shared K/V tiles)
    dim3 agrid(S_SEQ / 64, BH_CNT);
    yat_attn_wmma<<<agrid, 128, 0, stream>>>(Qn, Kn, Vbh, Obf);

    // 5) output projection -> fp32 d_out
    gemm_bf16_wmma<<<gemm_grid, 128, 0, stream>>>(Obf, Wob, bo, out, BS_ROWS, E_DIM, E_DIM);
}
